// ALIKED_27556510171209
// MI455X (gfx1250) — matrix operand order repaired and checked by reading, not compiled, not run on hardware
//
#include <hip/hip_runtime.h>
#include <cstdint>

#define NB   8
#define H    1536
#define W    2048
#define RAD  2
#define TOPK 1024
#define CAP  262144
#define TS   32
#define HALO 10
#define LW   52
#define NEGV (-1.0e30f)

typedef __attribute__((ext_vector_type(16))) _Float16 v16h_t;
typedef __attribute__((ext_vector_type(8)))  float    v8f_t;

// ---------------------------------------------------------------------------
// Kernel 1: fused 5-pool simple_nms on 32x32 tiles (halo 10), emit candidates
// ---------------------------------------------------------------------------
__global__ __launch_bounds__(256) void nms_collect_kernel(
    const float* __restrict__ scores,
    unsigned* __restrict__ cand_count,
    unsigned* __restrict__ cand_key,
    unsigned* __restrict__ cand_idx)
{
  __shared__ float S [LW*LW];
  __shared__ float M [LW*LW];
  __shared__ float A [LW*LW];
  __shared__ float Bp[LW*LW];
  __shared__ float C [LW*LW];

  const int tid = threadIdx.x;
  const int ox = blockIdx.x * TS - HALO;
  const int oy = blockIdx.y * TS - HALO;
  const int b  = blockIdx.z;
  const float* sb = scores + (size_t)b * H * W;

  for (int i = tid; i < LW*LW; i += 256) {
    int x = ox + (i % LW), y = oy + (i / LW);
    S[i] = (x >= 0 && x < W && y >= 0 && y < H) ? sb[(size_t)y * W + x] : NEGV;
    M[i] = 0.0f;
  }
  __syncthreads();

  auto hmax = [&](float* dst, const float* src, int x0, int x1, int y0, int y1) {
    const int wr = x1 - x0, n = wr * (y1 - y0);
    for (int i = tid; i < n; i += 256) {
      int x = x0 + i % wr, y = y0 + i / wr;
      const float* r = src + y * LW + x;
      dst[y*LW + x] = fmaxf(fmaxf(fmaxf(r[-2], r[-1]), fmaxf(r[0], r[1])), r[2]);
    }
  };
  auto vmaxAt = [&](const float* src, int j) -> float {
    return fmaxf(fmaxf(fmaxf(src[j-2*LW], src[j-LW]), fmaxf(src[j], src[j+LW])), src[j+2*LW]);
  };

  // max_mask0 = (S == P(S)), valid [2,50)^2
  hmax(A, S, 2, 50, 0, 52); __syncthreads();
  for (int i = tid; i < 48*48; i += 256) {
    int x = 2 + i % 48, y = 2 + i / 48, j = y*LW + x;
    float s = S[j];
    M[j] = (s != NEGV && s == vmaxAt(A, j)) ? 1.0f : 0.0f;
  }
  __syncthreads();

  // supp1 = P(mask0) > 0, valid [4,48)^2
  hmax(A, M, 4, 48, 2, 50); __syncthreads();
  for (int i = tid; i < 44*44; i += 256) {
    int x = 4 + i % 44, y = 4 + i / 44, j = y*LW + x;
    Bp[j] = (vmaxAt(A, j) > 0.0f) ? 1.0f : 0.0f;
  }
  __syncthreads();
  // ss1 = where(supp1, 0, S), valid [4,48)^2
  for (int i = tid; i < 44*44; i += 256) {
    int x = 4 + i % 44, y = 4 + i / 44, j = y*LW + x;
    C[j] = (Bp[j] > 0.0f) ? 0.0f : S[j];
  }
  __syncthreads();
  hmax(A, C, 6, 46, 4, 48); __syncthreads();
  // mask1 = mask0 | (new_max1 & ~supp1), valid [6,46)^2
  for (int i = tid; i < 40*40; i += 256) {
    int x = 6 + i % 40, y = 6 + i / 40, j = y*LW + x;
    bool nm = (S[j] != NEGV) && (C[j] == vmaxAt(A, j));
    M[j] = ((M[j] > 0.0f) || (nm && !(Bp[j] > 0.0f))) ? 1.0f : 0.0f;
  }
  __syncthreads();

  // supp2 = P(mask1) > 0, valid [8,44)^2
  hmax(A, M, 8, 44, 6, 46); __syncthreads();
  for (int i = tid; i < 36*36; i += 256) {
    int x = 8 + i % 36, y = 8 + i / 36, j = y*LW + x;
    Bp[j] = (vmaxAt(A, j) > 0.0f) ? 1.0f : 0.0f;
  }
  __syncthreads();
  for (int i = tid; i < 36*36; i += 256) {
    int x = 8 + i % 36, y = 8 + i / 36, j = y*LW + x;
    C[j] = (Bp[j] > 0.0f) ? 0.0f : S[j];
  }
  __syncthreads();
  hmax(A, C, 10, 42, 8, 44); __syncthreads();

  // final mask on output tile [10,42)^2; apply interior mask; emit candidates
  for (int i = tid; i < TS*TS; i += 256) {
    int x = 10 + i % TS, y = 10 + i / TS, j = y*LW + x;
    float s = S[j];
    bool nm = (s != NEGV) && (C[j] == vmaxAt(A, j));
    bool mk = (M[j] > 0.0f) || (nm && !(Bp[j] > 0.0f));
    int gx = ox + x, gy = oy + y;
    if (mk && s > 0.0f && gx >= RAD && gx < (W - RAD) && gy >= RAD && gy < (H - RAD)) {
      unsigned pos = atomicAdd(&cand_count[b], 1u);
      if (pos < CAP) {
        cand_key[(size_t)b*CAP + pos] = __float_as_uint(s);
        cand_idx[(size_t)b*CAP + pos] = (unsigned)(gy * W + gx);
      }
    }
  }
}

// ---------------------------------------------------------------------------
// Radix-select (4 byte passes, MSB first) over float-bit keys
// ---------------------------------------------------------------------------
__global__ __launch_bounds__(256) void hist_kernel(
    const unsigned* __restrict__ cand_key, const unsigned* __restrict__ cand_count,
    const unsigned* __restrict__ prefix, unsigned* __restrict__ hist, int pass)
{
  __shared__ unsigned hloc[256];
  const int b = blockIdx.y;
  hloc[threadIdx.x] = 0u;
  __syncthreads();
  unsigned cnt = cand_count[b]; if (cnt > CAP) cnt = CAP;
  const unsigned pfx = (pass == 0) ? 0u : prefix[b];
  const int shiftm = 32 - 8 * pass;
  const int shiftb = 24 - 8 * pass;
  for (unsigned i = blockIdx.x * blockDim.x + threadIdx.x; i < cnt; i += gridDim.x * blockDim.x) {
    unsigned key = cand_key[(size_t)b*CAP + i];
    unsigned long long k64 = key;
    bool match = (pass == 0) || ((unsigned)(k64 >> shiftm) == pfx);
    if (match) atomicAdd(&hloc[(key >> shiftb) & 0xFFu], 1u);
  }
  __syncthreads();
  if (hloc[threadIdx.x]) atomicAdd(&hist[b*256 + threadIdx.x], hloc[threadIdx.x]);
}

__global__ void select_kernel(const unsigned* __restrict__ hist,
                              unsigned* __restrict__ prefix, int* __restrict__ need, int pass)
{
  int b = threadIdx.x;
  if (b >= NB) return;
  int need_in = (pass == 0) ? TOPK : need[b];
  unsigned pfx = (pass == 0) ? 0u : prefix[b];
  int cum = 0, bin = 0;
  for (int i = 255; i >= 0; --i) {
    int c = (int)hist[b*256 + i];
    if (cum + c >= need_in) { bin = i; need[b] = need_in - cum; break; }
    cum += c;
  }
  prefix[b] = (pfx << 8) | (unsigned)bin;
}

__global__ __launch_bounds__(256) void collect_kernel(
    const unsigned* __restrict__ cand_key, const unsigned* __restrict__ cand_idx,
    const unsigned* __restrict__ cand_count, const unsigned* __restrict__ prefix,
    unsigned* __restrict__ topcount, unsigned long long* __restrict__ topbuf)
{
  const int b = blockIdx.y;
  unsigned cnt = cand_count[b]; if (cnt > CAP) cnt = CAP;
  const unsigned T = prefix[b];
  for (unsigned i = blockIdx.x * blockDim.x + threadIdx.x; i < cnt; i += gridDim.x * blockDim.x) {
    unsigned key = cand_key[(size_t)b*CAP + i];
    if (key >= T) {
      unsigned pos = atomicAdd(&topcount[b], 1u);
      if (pos < 2048u) {
        unsigned idx = cand_idx[(size_t)b*CAP + i];
        topbuf[(size_t)b*2048 + pos] =
            ((unsigned long long)key << 32) | (unsigned long long)(0xFFFFFFFFu - idx);
      }
    }
  }
}

// Bitonic sort 2048 composites descending => exact jax.lax.top_k order
__global__ __launch_bounds__(1024) void sort_kernel(
    const unsigned long long* __restrict__ topbuf,
    const unsigned* __restrict__ topcount, int* __restrict__ topk_idx)
{
  __shared__ unsigned long long arr[2048];
  const int b = blockIdx.x, t = threadIdx.x;
  unsigned cnt = topcount[b]; if (cnt > 2048u) cnt = 2048u;
  for (int i = t; i < 2048; i += 1024)
    arr[i] = (i < (int)cnt) ? topbuf[(size_t)b*2048 + i] : 0ull;
  __syncthreads();
  for (int k = 2; k <= 2048; k <<= 1) {
    for (int j = k >> 1; j > 0; j >>= 1) {
      for (int m = t; m < 2048; m += 1024) {
        int l = m ^ j;
        if (l > m) {
          unsigned long long a0 = arr[m], a1 = arr[l];
          bool descBlock = ((m & k) == 0);
          bool sw = descBlock ? (a0 < a1) : (a0 > a1);
          if (sw) { arr[m] = a1; arr[l] = a0; }
        }
      }
      __syncthreads();
    }
  }
  if (t < TOPK) {
    unsigned long long e = arr[t];
    unsigned idx = 0xFFFFFFFFu - (unsigned)(e & 0xFFFFFFFFull);
    if (e == 0ull) idx = (unsigned)(RAD * W + RAD);
    topk_idx[b*TOPK + t] = (int)idx;
  }
}

// ---------------------------------------------------------------------------
// Kernel 5: per-keypoint softmax refinement. Moments Σe·[1,gx,gy,gx²+gy²]
// computed with V_WMMA_F32_16X16X32_F16 (A = exponentials, B = exact grid).
// ---------------------------------------------------------------------------
__global__ __launch_bounds__(256) void refine_kernel(
    const float* __restrict__ scores, const int* __restrict__ topk_idx,
    float* __restrict__ out)
{
  __shared__ float mom[8][32][8];
  const int b    = blockIdx.y;
  const int lane = threadIdx.x & 31;
  const int wave = threadIdx.x >> 5;
  const int m    = lane & 15;     // keypoint-in-wave
  const int half = lane >> 4;     // which K-half of A this lane supplies
  const int k    = blockIdx.x * 128 + wave * 16 + m;

  int idx = topk_idx[b*TOPK + k];
  int ys = idx / W, xs = idx % W;
  ys = min(max(ys, RAD), H - 1 - RAD);
  xs = min(max(xs, RAD), W - 1 - RAD);
  const float* sb = scores + (size_t)b * H * W;

  float p[25], mx = -1.0e30f;
#pragma unroll
  for (int n = 0; n < 25; ++n) {
    p[n] = sb[(size_t)(ys - 2 + n / 5) * W + (xs - 2 + n % 5)];
    mx = fmaxf(mx, p[n]);
  }
  float e[25];
#pragma unroll
  for (int n = 0; n < 25; ++n) e[n] = expf((p[n] - mx) * 10.0f);  // 1/TEMP

  // A (16x32 f16) per ISA layout: lane<16 -> K{0..7,16..23}; lane>=16 -> K{8..15,24..31}
  v16h_t av;
#pragma unroll
  for (int j = 0; j < 8; ++j) {
    float lo = half ? e[8 + j] : e[j];
    float hi = half ? ((j == 0) ? e[24] : 0.0f) : e[16 + j];
    av[j]     = (_Float16)lo;
    av[8 + j] = (_Float16)hi;
  }
  // B (32x16 f16): column = lane%16 => [1, gx, gy, gx^2+gy^2, 0...]; lane half picks K range
  v16h_t bv;
  const int col = m, kb = half * 16;
#pragma unroll
  for (int j = 0; j < 16; ++j) {
    int kk = (j < 8) ? (half * 8 + j) : (16 + half * 8 + (j - 8));
    float gx = (float)(kk - (kk / 5) * 5) - 2.0f;
    float gy = (float)(kk / 5) - 2.0f;
    float v = 0.0f;
    v = (col == 0) ? 1.0f : v;
    v = (col == 1) ? gx   : v;
    v = (col == 2) ? gy   : v;
    v = (col == 3) ? (gx*gx + gy*gy) : v;
    v = (kk < 25) ? v : 0.0f;
    bv[j] = (_Float16)v;
  }
  v8f_t cz = {0.f, 0.f, 0.f, 0.f, 0.f, 0.f, 0.f, 0.f};
  v8f_t d = __builtin_amdgcn_wmma_f32_16x16x32_f16(false, av, false, bv,
                                                   (short)0, cz, false, false);
#pragma unroll
  for (int j = 0; j < 8; ++j) mom[wave][lane][j] = d[j];
  __syncthreads();

  if (lane < 16) {
    // D[M][N] lives at lane = N + 16*(M>=8), vgpr = M%8
    const int hb = (m >= 8) ? 16 : 0, r = m & 7;
    float Sv = mom[wave][hb + 0][r];
    float Sx = mom[wave][hb + 1][r];
    float Sy = mom[wave][hb + 2][r];
    float Sq = mom[wave][hb + 3][r];
    float rx = Sx / Sv, ry = Sy / Sv;
    float disp = (Sq - (Sx * Sx + Sy * Sy) / Sv) / (4.0f * Sv);  // R^2 = 4
    float kx = ((float)xs + rx) / (float)(W - 1) * 2.0f - 1.0f;
    float ky = ((float)ys + ry) / (float)(H - 1) * 2.0f - 1.0f;

    float px = (kx + 1.0f) * 0.5f * (float)(W - 1);
    float py = (ky + 1.0f) * 0.5f * (float)(H - 1);
    float x0 = floorf(px), y0 = floorf(py);
    float wx = px - x0, wy = py - y0;
    int x0i = min(max((int)x0, 0), W - 1), x1i = min(x0i + 1, W - 1);
    int y0i = min(max((int)y0, 0), H - 1), y1i = min(y0i + 1, H - 1);
    float v00 = sb[(size_t)y0i * W + x0i], v01 = sb[(size_t)y0i * W + x1i];
    float v10 = sb[(size_t)y1i * W + x0i], v11 = sb[(size_t)y1i * W + x1i];
    float sc = v00 * (1-wx)*(1-wy) + v01 * wx*(1-wy) + v10 * (1-wx)*wy + v11 * wx*wy;

    const int o = b * TOPK + k;
    out[(size_t)o * 2 + 0] = kx;
    out[(size_t)o * 2 + 1] = ky;
    out[NB * TOPK * 2 + o] = disp;
    out[NB * TOPK * 3 + o] = sc;
  }
}

// ---------------------------------------------------------------------------
// Workspace layout (bytes):
//   0      cand_count[8]    32 prefix[8]    64 need[8]    96 topcount[8]
//   256    hist[8][256]                       (8192 B)
//   8704   topbuf[8][2048] u64                (131072 B)
//   139776 topk_idx[8][1024] int              (32768 B)
//   262144 cand_key[8][CAP] u32               (8 MiB)
//   +8MiB  cand_idx[8][CAP] u32               (8 MiB)   total ~16.3 MiB
// ---------------------------------------------------------------------------
extern "C" void kernel_launch(void* const* d_in, const int* in_sizes, int n_in,
                              void* d_out, int out_size, void* d_ws, size_t ws_size,
                              hipStream_t stream)
{
  (void)in_sizes; (void)n_in; (void)out_size; (void)ws_size;
  const float* scores = (const float*)d_in[0];
  char* ws = (char*)d_ws;

  unsigned* cand_count = (unsigned*)(ws + 0);
  unsigned* prefix     = (unsigned*)(ws + 32);
  int*      need       = (int*)     (ws + 64);
  unsigned* topcount   = (unsigned*)(ws + 96);
  unsigned* hist       = (unsigned*)(ws + 256);
  unsigned long long* topbuf = (unsigned long long*)(ws + 8704);
  int*      topk_idx   = (int*)     (ws + 139776);
  unsigned* cand_key   = (unsigned*)(ws + 262144);
  unsigned* cand_idx   = (unsigned*)(ws + 262144 + (size_t)NB * CAP * 4);
  float* out = (float*)d_out;

  hipMemsetAsync(ws, 0, 8448, stream);  // counters + state + hist

  nms_collect_kernel<<<dim3(W / TS, H / TS, NB), 256, 0, stream>>>(
      scores, cand_count, cand_key, cand_idx);

  for (int pass = 0; pass < 4; ++pass) {
    if (pass) hipMemsetAsync(ws + 256, 0, 8192, stream);
    hist_kernel<<<dim3(64, NB), 256, 0, stream>>>(cand_key, cand_count, prefix, hist, pass);
    select_kernel<<<1, 8, 0, stream>>>(hist, prefix, need, pass);
  }

  collect_kernel<<<dim3(64, NB), 256, 0, stream>>>(
      cand_key, cand_idx, cand_count, prefix, topcount, topbuf);
  sort_kernel<<<NB, 1024, 0, stream>>>(topbuf, topcount, topk_idx);
  refine_kernel<<<dim3(8, NB), 256, 0, stream>>>(scores, topk_idx, out);
}
